// HeterogeneousDirectedGNN_75118978007323
// MI455X (gfx1250) — matrix operand order repaired and checked by reading, not compiled, not run on hardware
//
#include <hip/hip_runtime.h>
#include <cstdint>
#include <cstddef>

// ---------------------------------------------------------------------------
// Heterogeneous directed GNN forward for MI455X (gfx1250, wave32, WMMA).
// GEMMs: bf16 WMMA 16x16x32 with fp32 accumulate. Each wave computes a
// 16x128 output strip: one A fragment (converted fp32->bf16 via packed cvt)
// feeds 8 WMMAs per K-step; all 8 B fragments are staged first so the WMMAs
// issue back-to-back behind a single load wait. Weights pre-transposed bf16.
// Edge attention: per-edge score waves + atomic segment-max / segment-sum.
// ---------------------------------------------------------------------------

#define NGN  32768     // nodes per type (NG == NPK)
#define EE   262144    // edges per direction
#define HIDD 128
#define NHEAD 4
// SCALE = DH^-0.5 = 1/sqrt(32)
#define SCALE_F 0.17677669529663687f

typedef __attribute__((ext_vector_type(16))) __bf16 v16bf;
typedef __attribute__((ext_vector_type(4)))  __bf16 v4bf;
typedef __attribute__((ext_vector_type(8)))  float  v8f;
typedef __attribute__((ext_vector_type(4)))  float  v4f;

__device__ __forceinline__ unsigned short f2bf(float f) {
  unsigned int u = __float_as_uint(f);
  unsigned int r = u + 0x7FFFu + ((u >> 16) & 1u);   // round-to-nearest-even
  return (unsigned short)(r >> 16);
}

__device__ __forceinline__ void atomicMaxF(float* addr, float v) {
  if (v >= 0.0f) atomicMax((int*)addr, __float_as_int(v));
  else           atomicMin((unsigned int*)addr, __float_as_uint(v));
}

// ---------------------------------------------------------------------------
// GEMM: C[M x N] = A[M x K] @ W[K x N] + bias (+bias2) (+add) (relu)
// WT = W transposed to [N x K] bf16. One wave per 16-row x 128-col strip.
// ---------------------------------------------------------------------------
template <int KV, bool HASB2, bool HASADD, bool RELU>
__global__ __launch_bounds__(256) void gemm_wmma_t(
    const float* __restrict__ A, const unsigned short* __restrict__ WT,
    const float* __restrict__ bias, const float* __restrict__ bias2,
    const float* __restrict__ add, float* __restrict__ C, int M, int N)
{
  int wave = (blockIdx.x * blockDim.x + threadIdx.x) >> 5;
  int lane = threadIdx.x & 31;
  int nblk = N >> 7;                    // number of 128-column strips
  int tm = wave / nblk;
  int cb = wave - tm * nblk;
  if ((tm << 4) >= M) return;           // uniform per wave
  int half = lane >> 4;                 // 0: lanes 0-15, 1: lanes 16-31
  int r    = lane & 15;

  const float*          arow  = A  + (size_t)((tm << 4) + r) * KV;
  const unsigned short* bbase = WT + (size_t)((cb << 7) + r) * KV;

  v8f acc[8];
#pragma unroll
  for (int t = 0; t < 8; t++) {
    v8f z = {0.f, 0.f, 0.f, 0.f, 0.f, 0.f, 0.f, 0.f};
    acc[t] = z;
  }

#pragma unroll
  for (int k0 = 0; k0 < KV; k0 += 32) {
    // Stage all 8 B fragments for this K-step (one clause, one wait).
    // B fragment (32x16 bf16): lanes 0-15 hold K=k0+[0..15] of column
    // cb*128 + t*16 + r; lanes 16-31 hold K=k0+16+[0..15]. Contiguous in WT.
    v16bf bv[8];
#pragma unroll
    for (int t = 0; t < 8; t++)
      bv[t] = *(const v16bf*)(bbase + (size_t)t * 16 * KV + k0 + (half << 4));

    // A fragment (16x32 bf16 layout): half=0 holds K = k0+[0..15];
    // half=1 holds K = k0+16+[0..15] (matches B).
    int kaA = k0 + (half << 4);
    v4f a0 = *(const v4f*)(arow + kaA);
    v4f a1 = *(const v4f*)(arow + kaA + 4);
    v4f a2 = *(const v4f*)(arow + kaA + 8);
    v4f a3 = *(const v4f*)(arow + kaA + 12);
    union { v16bf v; v4bf q[4]; } af;
    af.q[0] = __builtin_convertvector(a0, v4bf);
    af.q[1] = __builtin_convertvector(a1, v4bf);
    af.q[2] = __builtin_convertvector(a2, v4bf);
    af.q[3] = __builtin_convertvector(a3, v4bf);

#pragma unroll
    for (int t = 0; t < 8; t++)
      acc[t] = __builtin_amdgcn_wmma_f32_16x16x32_bf16(
          false, af.v, false, bv[t], (short)0, acc[t], false, false);
  }

  int ncol0 = (cb << 7) + r;
#pragma unroll
  for (int t = 0; t < 8; t++) {
    int n = ncol0 + (t << 4);
    float bb = bias[n];
    if (HASB2) bb += bias2[n];
#pragma unroll
    for (int g = 0; g < 8; g++) {
      int m = (tm << 4) + (half << 3) + g;   // D layout: VGPR g -> M=g / M=g+8
      float val = acc[t][g] + bb;
      if (HASADD) val += add[(size_t)m * N + n];
      if (RELU)   val = fmaxf(val, 0.f);
      C[(size_t)m * N + n] = val;
    }
  }
}

// W[K x N] fp32 -> WT[N x K] bf16
__global__ void wt_convert(const float* __restrict__ W,
                           unsigned short* __restrict__ WT, int K, int N) {
  int idx = blockIdx.x * blockDim.x + threadIdx.x;
  if (idx >= K * N) return;
  int k = idx / N, n = idx - k * N;
  WT[(size_t)n * K + k] = f2bf(W[idx]);
}

__global__ void fill_f32(float* __restrict__ p, float v, int n) {
  int i = blockIdx.x * blockDim.x + threadIdx.x;
  if (i < n) p[i] = v;
}

// One wave per edge: s[e][h] = SCALE * ew[e] * <q[tgt], k[src]>_head, plus
// atomic segment max into mbuf[tgt][h].
__global__ __launch_bounds__(256) void edge_scores(
    const float* __restrict__ q, const float* __restrict__ k,
    const int* __restrict__ src, const int* __restrict__ tgt,
    const float* __restrict__ ew, float* __restrict__ s,
    float* __restrict__ mbuf, int E)
{
  int e    = (blockIdx.x * blockDim.x + threadIdx.x) >> 5;
  int lane = threadIdx.x & 31;
  if (e >= E) return;
  int ts = tgt[e], ss = src[e];
  v4f qv = *(const v4f*)(q + (size_t)ts * HIDD + lane * 4);
  v4f kv = *(const v4f*)(k + (size_t)ss * HIDD + lane * 4);
  float p = qv.x * kv.x + qv.y * kv.y + qv.z * kv.z + qv.w * kv.w;
  // reduce within each 8-lane head group (head = lane>>3, 32 dims per head)
  p += __shfl_xor(p, 1);
  p += __shfl_xor(p, 2);
  p += __shfl_xor(p, 4);
  if ((lane & 7) == 0) {
    int h = lane >> 3;
    float sv = p * SCALE_F * ew[e];
    s[(size_t)e * NHEAD + h] = sv;
    atomicMaxF(mbuf + (size_t)ts * NHEAD + h, sv);
  }
}

// One wave per edge: ex = exp(s - m[tgt]); d[tgt][h] += ex;
// agg[tgt][:] += ex * v[src][:]  (un-normalized; divided by d later)
__global__ __launch_bounds__(256) void edge_accum(
    const float* __restrict__ s, const float* __restrict__ v,
    const int* __restrict__ src, const int* __restrict__ tgt,
    const float* __restrict__ mbuf, float* __restrict__ dbuf,
    float* __restrict__ agg, int E)
{
  int e    = (blockIdx.x * blockDim.x + threadIdx.x) >> 5;
  int lane = threadIdx.x & 31;
  if (e >= E) return;
  int ts = tgt[e], ss = src[e];
  int h = lane >> 3;
  float ex = __expf(s[(size_t)e * NHEAD + h] - mbuf[(size_t)ts * NHEAD + h]);
  if ((lane & 7) == 0) atomicAdd(dbuf + (size_t)ts * NHEAD + h, ex);
  v4f vv = *(const v4f*)(v + (size_t)ss * HIDD + lane * 4);
  float* ap = agg + (size_t)ts * HIDD + lane * 4;
  atomicAdd(ap + 0, ex * vv.x);
  atomicAdd(ap + 1, ex * vv.y);
  atomicAdd(ap + 2, ex * vv.z);
  atomicAdd(ap + 3, ex * vv.w);
}

__global__ void norm_agg(float* __restrict__ agg, const float* __restrict__ dbuf,
                         int total) {
  int i = blockIdx.x * blockDim.x + threadIdx.x;
  if (i >= total) return;
  int node = i >> 7;
  int c = i & 127;
  float dd = dbuf[node * NHEAD + (c >> 5)];
  agg[i] = dd > 0.f ? agg[i] / dd : 0.f;
}

// One wave per 128-dim row: LayerNorm with wave32 shuffle reductions.
__global__ __launch_bounds__(256) void layernorm_k(
    const float* __restrict__ x, const float* __restrict__ g,
    const float* __restrict__ b, float* __restrict__ y, int Nrows)
{
  int row  = (blockIdx.x * blockDim.x + threadIdx.x) >> 5;
  int lane = threadIdx.x & 31;
  if (row >= Nrows) return;
  v4f xv = *(const v4f*)(x + (size_t)row * HIDD + lane * 4);
  float sm = xv.x + xv.y + xv.z + xv.w;
  for (int o = 1; o < 32; o <<= 1) sm += __shfl_xor(sm, o);
  float mu = sm * (1.f / 128.f);
  v4f dv = {xv.x - mu, xv.y - mu, xv.z - mu, xv.w - mu};
  float sq = dv.x * dv.x + dv.y * dv.y + dv.z * dv.z + dv.w * dv.w;
  for (int o = 1; o < 32; o <<= 1) sq += __shfl_xor(sq, o);
  float inv = rsqrtf(sq * (1.f / 128.f) + 1e-5f);
  v4f gv = *(const v4f*)(g + lane * 4);
  v4f bv = *(const v4f*)(b + lane * 4);
  v4f out = {dv.x * inv * gv.x + bv.x, dv.y * inv * gv.y + bv.y,
             dv.z * inv * gv.z + bv.z, dv.w * inv * gv.w + bv.w};
  *(v4f*)(y + (size_t)row * HIDD + lane * 4) = out;
}

// ---------------------------------------------------------------------------
// Host orchestration.
// Input leaf order (setup_inputs() insertion order, recursively flattened):
//  0 x_gene  1 x_peak  2 e_g2p(int)  3 e_p2g(int)  4 ew_g2p  5 ew_p2g
//  6 emb.gene  7 emb.peak
//  8/9 in_proj.gene.W/b  10/11 in_proj.peak.W/b
//  12/13 out_proj.gene.W/b  14/15 out_proj.peak.W/b
//  per layer l (base = 16 + 36*l):
//    +0..+7  edge.g2p q.W,q.b,k.W,k.b,v.W,v.b,o.W,o.b
//    +8..+15 edge.p2g (same order)
//    +16..+23 self gene v.W,v.b,o.W,o.b ; peak v.W,v.b,o.W,o.b
//    +24..+31 ffn gene l1.W,l1.b,l2.W,l2.b ; peak same
//    +32..+35 ln gene g,b ; peak g,b
// ---------------------------------------------------------------------------
extern "C" void kernel_launch(void* const* d_in, const int* in_sizes, int n_in,
                              void* d_out, int out_size, void* d_ws, size_t ws_size,
                              hipStream_t stream) {
  (void)in_sizes; (void)n_in; (void)out_size; (void)ws_size;
  auto F = [&](int i) { return (const float*)d_in[i]; };
  const float* x_gene = F(0);
  const float* x_peak = F(1);
  const int*   eg     = (const int*)d_in[2];
  const int*   ep     = (const int*)d_in[3];
  const float* ewg    = F(4);
  const float* ewp    = F(5);

  // --- workspace carve ---
  uintptr_t wp = (uintptr_t)d_ws;
  auto carve = [&](size_t bytes) -> void* {
    void* p = (void*)wp;
    wp += (bytes + 255) & ~(size_t)255;
    return p;
  };
  float* hg  = (float*)carve((size_t)NGN * HIDD * 4);
  float* hp  = (float*)carve((size_t)NGN * HIDD * 4);
  float* qb  = (float*)carve((size_t)NGN * HIDD * 4);   // also self-V temp
  float* kb  = (float*)carve((size_t)NGN * HIDD * 4);   // also comb temp
  float* vb  = (float*)carve((size_t)NGN * HIDD * 4);   // also ffn-out temp
  float* sb  = (float*)carve((size_t)EE * NHEAD * 4);
  float* mb  = (float*)carve((size_t)NGN * NHEAD * 4);
  float* db  = (float*)carve((size_t)NGN * NHEAD * 4);
  float* agg = (float*)carve((size_t)NGN * HIDD * 4);
  float* upg = (float*)carve((size_t)NGN * HIDD * 4);
  float* upp = (float*)carve((size_t)NGN * HIDD * 4);
  float* t1  = (float*)carve((size_t)NGN * 2 * HIDD * 4);

  unsigned short* cw_in[2];
  unsigned short* cw_out[2];
  unsigned short* cw_edge[2][8];
  unsigned short* cw_self[2][4];
  unsigned short* cw_ffn[2][4];
  for (int i = 0; i < 2; i++) cw_in[i]  = (unsigned short*)carve((size_t)512 * HIDD * 2);
  for (int i = 0; i < 2; i++) cw_out[i] = (unsigned short*)carve((size_t)HIDD * HIDD * 2);
  for (int l = 0; l < 2; l++) {
    for (int i = 0; i < 8; i++) cw_edge[l][i] = (unsigned short*)carve((size_t)HIDD * HIDD * 2);
    for (int i = 0; i < 4; i++) cw_self[l][i] = (unsigned short*)carve((size_t)HIDD * HIDD * 2);
    for (int i = 0; i < 4; i++) cw_ffn[l][i]  = (unsigned short*)carve((size_t)HIDD * 2 * HIDD * 2);
  }

  auto conv = [&](int idx, unsigned short* dst, int K, int N) {
    int n = K * N;
    wt_convert<<<(n + 255) / 256, 256, 0, stream>>>(F(idx), dst, K, N);
  };
  conv(8,  cw_in[0],  512, HIDD);
  conv(10, cw_in[1],  512, HIDD);
  conv(12, cw_out[0], HIDD, HIDD);
  conv(14, cw_out[1], HIDD, HIDD);
  for (int l = 0; l < 2; l++) {
    int base = 16 + 36 * l;
    const int eoff[8] = {0, 2, 4, 6, 8, 10, 12, 14};
    for (int i = 0; i < 8; i++) conv(base + eoff[i], cw_edge[l][i], HIDD, HIDD);
    const int soff[4] = {16, 18, 20, 22};
    for (int i = 0; i < 4; i++) conv(base + soff[i], cw_self[l][i], HIDD, HIDD);
    conv(base + 24, cw_ffn[l][0], HIDD, 2 * HIDD);
    conv(base + 26, cw_ffn[l][1], 2 * HIDD, HIDD);
    conv(base + 28, cw_ffn[l][2], HIDD, 2 * HIDD);
    conv(base + 30, cw_ffn[l][3], 2 * HIDD, HIDD);
  }

  // Dispatch to the exact template instantiations the pipeline uses.
  auto gemm = [&](const float* A, const unsigned short* WT, const float* bias,
                  const float* bias2, const float* add, float* C,
                  int M, int K, int N, int relu) {
    int waves = (M >> 4) * (N >> 7);
    dim3 gr(waves / 8), bl(256);
    if (K == 512) {          // in_proj (+embedding as bias2)
      gemm_wmma_t<512, true, false, false><<<gr, bl, 0, stream>>>(A, WT, bias, bias2, add, C, M, N);
    } else if (K == 256) {   // ffn l2 (+residual)
      gemm_wmma_t<256, false, true, false><<<gr, bl, 0, stream>>>(A, WT, bias, bias2, add, C, M, N);
    } else if (relu) {       // ffn l1
      gemm_wmma_t<128, false, false, true><<<gr, bl, 0, stream>>>(A, WT, bias, bias2, add, C, M, N);
    } else if (add) {        // self out-proj (+up)
      gemm_wmma_t<128, false, true, false><<<gr, bl, 0, stream>>>(A, WT, bias, bias2, add, C, M, N);
    } else {                 // plain projections
      gemm_wmma_t<128, false, false, false><<<gr, bl, 0, stream>>>(A, WT, bias, bias2, add, C, M, N);
    }
  };
  auto fill = [&](float* p, float v, int n) {
    fill_f32<<<(n + 255) / 256, 256, 0, stream>>>(p, v, n);
  };

  const float NEG_INF = -__builtin_inff();

  // in_proj + type embedding (broadcast vector == second bias)
  gemm(x_gene, cw_in[0], F(9),  F(6), nullptr, hg, NGN, 512, HIDD, 0);
  gemm(x_peak, cw_in[1], F(11), F(7), nullptr, hp, NGN, 512, HIDD, 0);

  auto edge_attn = [&](const float* hsrc, const float* htgt, const int* edges,
                       const float* ew, unsigned short* const* W4,
                       const float* bq, const float* bk, const float* bvv,
                       const float* bo, float* up) {
    const int* src = edges;
    const int* tgt = edges + EE;
    gemm(htgt, W4[0], bq,  nullptr, nullptr, qb, NGN, HIDD, HIDD, 0);
    gemm(hsrc, W4[1], bk,  nullptr, nullptr, kb, NGN, HIDD, HIDD, 0);
    gemm(hsrc, W4[2], bvv, nullptr, nullptr, vb, NGN, HIDD, HIDD, 0);
    fill(mb, NEG_INF, NGN * NHEAD);
    fill(db, 0.f, NGN * NHEAD);
    fill(agg, 0.f, NGN * HIDD);
    edge_scores<<<EE / 8, 256, 0, stream>>>(qb, kb, src, tgt, ew, sb, mb, EE);
    edge_accum<<<EE / 8, 256, 0, stream>>>(sb, vb, src, tgt, mb, db, agg, EE);
    norm_agg<<<(NGN * HIDD + 255) / 256, 256, 0, stream>>>(agg, db, NGN * HIDD);
    gemm(agg, W4[3], bo, nullptr, nullptr, up, NGN, HIDD, HIDD, 0);
  };

  for (int l = 0; l < 2; l++) {
    int base = 16 + 36 * l;
    // g2p: gene -> peak  (updates peak)
    edge_attn(hg, hp, eg, ewg, &cw_edge[l][0],
              F(base + 1), F(base + 3), F(base + 5), F(base + 7), upp);
    // p2g: peak -> gene  (updates gene)
    edge_attn(hp, hg, ep, ewp, &cw_edge[l][4],
              F(base + 9), F(base + 11), F(base + 13), F(base + 15), upg);

    for (int nt = 0; nt < 2; nt++) {   // 0 = gene, 1 = peak
      float* h  = nt == 0 ? hg : hp;
      float* up = nt == 0 ? upg : upp;
      unsigned short* wv = cw_self[l][nt * 2 + 0];
      unsigned short* wo = cw_self[l][nt * 2 + 1];
      const float* bvv = F(base + 17 + nt * 4);
      const float* boo = F(base + 19 + nt * 4);
      unsigned short* w1 = cw_ffn[l][nt * 2 + 0];
      unsigned short* w2 = cw_ffn[l][nt * 2 + 1];
      const float* b1  = F(base + 25 + nt * 4);
      const float* b2  = F(base + 27 + nt * 4);
      const float* lng = F(base + 32 + nt * 2);
      const float* lnb = F(base + 33 + nt * 2);
      float* tsv = qb;   // free after edge phase
      float* comb = kb;
      float* t2 = vb;
      // self-attn (seq_len==1): out_proj(value_proj(h)); comb = sa + up
      gemm(h,   wv, bvv, nullptr, nullptr, tsv,  NGN, HIDD, HIDD, 0);
      gemm(tsv, wo, boo, nullptr, up,      comb, NGN, HIDD, HIDD, 0);
      // FFN with residual, then LayerNorm into h
      gemm(comb, w1, b1, nullptr, nullptr, t1, NGN, HIDD, 2 * HIDD, 1);
      gemm(t1,   w2, b2, nullptr, h,       t2, NGN, 2 * HIDD, HIDD, 0);
      layernorm_k<<<NGN / 8, 256, 0, stream>>>(t2, lng, lnb, h, NGN);
    }
  }

  float* outp = (float*)d_out;
  gemm(hg, cw_out[0], F(13), nullptr, nullptr, outp,                      NGN, HIDD, HIDD, 0);
  gemm(hp, cw_out[1], F(15), nullptr, nullptr, outp + (size_t)NGN * HIDD, NGN, HIDD, HIDD, 0);
}
